// MultiHeadedAttention_47940424958053
// MI455X (gfx1250) — matrix operand order repaired and checked by reading, not compiled, not run on hardware
//
#include <hip/hip_runtime.h>

// MultiHeadedAttention for MI455X (gfx1250, wave32, WMMA).
// B=4, S=2048, D_MODEL=1024, H=16, Dk=64.
// v2: bf16 weight pre-conversion (W reused 64x -> convert once),
//     2 M-subtiles per wave in the projection (8 WMMA per B-fragment set),
//     64-wide key blocks in attention (16 WMMA per softmax round),
//     global_prefetch_b8 on streaming operands.

#define DEV __device__ __forceinline__

typedef __bf16 bf16_t;
typedef bf16_t v16bf_t __attribute__((ext_vector_type(16)));
typedef bf16_t v8bf_t  __attribute__((ext_vector_type(8)));
typedef float  v8f_t   __attribute__((ext_vector_type(8)));
typedef float  v4f_t   __attribute__((ext_vector_type(4)));

static constexpr int S_LEN  = 2048;
static constexpr int DMODEL = 1024;
static constexpr int HEADS  = 16;
static constexpr int DK     = 64;

DEV v8f_t wmma_bf16(v16bf_t a, v16bf_t b, v8f_t c) {
  // D = A(16x32 bf16) * B(32x16 bf16) + C(16x16 f32)
  return __builtin_amdgcn_wmma_f32_16x16x32_bf16(
      /*neg_a=*/false, a, /*neg_b=*/false, b,
      /*c_mod=*/(short)0, c, /*reuse_a=*/false, /*reuse_b=*/false);
}

// ---- Fragment loaders (ISA 7.12.2 16-bit layouts) -------------------------
// A (16x32, MxK): lane<16 -> row=lane,   elems 0..7 = K[kb..kb+7],   8..15 = K[kb+16..kb+23]
//                 lane>=16-> row=lane-16,elems 0..7 = K[kb+8..kb+15],8..15 = K[kb+24..kb+31]
DEV v16bf_t frag_a_bf16(const bf16_t* __restrict__ base, int row, int ld,
                        int kb, int lane) {
  const bf16_t* r = base + (size_t)row * ld + kb + ((lane & 16) ? 8 : 0);
  v8bf_t lo = *(const v8bf_t*)(r);
  v8bf_t hi = *(const v8bf_t*)(r + 16);
  v16bf_t f;
#pragma unroll
  for (int i = 0; i < 8; ++i) { f[i] = lo[i]; f[i + 8] = hi[i]; }
  return f;
}

// B (32x16, KxN), source stored as B^T rows (row n, contiguous along K):
// lane<16 -> col n, elems 0..7 = K[kb..kb+7], 8..15 = K[kb+16..kb+23]; lane>=16 -> K[kb+8..kb+15], K[kb+24..kb+31]
DEV v16bf_t frag_b_bf16(const bf16_t* __restrict__ base, int row, int ld,
                        int kb, int lane) {
  const bf16_t* r = base + (size_t)row * ld + kb + ((lane & 16) ? 8 : 0);
  v8bf_t lo = *(const v8bf_t*)(r);
  v8bf_t hi = *(const v8bf_t*)(r + 16);
  v16bf_t f;
#pragma unroll
  for (int i = 0; i < 8; ++i) { f[i] = lo[i]; f[i + 8] = hi[i]; }
  return f;
}

// A fragment sourced from fp32 (convert to bf16 on the fly).
DEV v16bf_t frag_a_f32(const float* __restrict__ base, int row, int ld,
                       int kb, int lane) {
  const float* r = base + (size_t)row * ld + kb + ((lane & 16) ? 8 : 0);
  v4f_t a0 = *(const v4f_t*)(r);
  v4f_t a1 = *(const v4f_t*)(r + 4);
  v4f_t a2 = *(const v4f_t*)(r + 16);
  v4f_t a3 = *(const v4f_t*)(r + 20);
  v16bf_t f;
#pragma unroll
  for (int i = 0; i < 4; ++i) {
    f[i]      = (bf16_t)a0[i];
    f[4 + i]  = (bf16_t)a1[i];
    f[8 + i]  = (bf16_t)a2[i];
    f[12 + i] = (bf16_t)a3[i];
  }
  return f;
}

// ---- Kernel 0: elementwise fp32 -> bf16 (for the 3 weight matrices) -------
__global__ __launch_bounds__(256) void cvt_f32_bf16_kernel(
    const float* __restrict__ src, bf16_t* __restrict__ dst, int n) {
  const int i = (blockIdx.x * 256 + threadIdx.x) * 8;
  if (i + 8 <= n) {
    v4f_t a = *(const v4f_t*)(src + i);
    v4f_t b = *(const v4f_t*)(src + i + 4);
    v8bf_t o;
#pragma unroll
    for (int k = 0; k < 4; ++k) {
      o[k]     = (bf16_t)a[k];
      o[4 + k] = (bf16_t)b[k];
    }
    *(v8bf_t*)(dst + i) = o;
  }
}

// ---- Kernel 1: P = X @ Wbf^T + b, write bf16 head-major -------------------
// grid = (32 Mtiles of 256 rows, 16 heads), block = 256 (8 waves).
// Each wave: 32 rows x 64 cols (2 M-subtiles x 4 N-tiles = 8 WMMA / k-step).
// transposed==0 -> out[b][h][s][dk] ; transposed==1 -> out[b][h][dk][s]
__global__ __launch_bounds__(256) void proj_bf16_kernel(
    const float* __restrict__ X, const bf16_t* __restrict__ Wbf,
    const float* __restrict__ bias, bf16_t* __restrict__ out, int transposed) {
  const int lane = threadIdx.x & 31;
  const int wave = threadIdx.x >> 5;
  const int m0 = blockIdx.x * 256 + wave * 32;  // row of X (over B*S = 8192)
  const int h  = blockIdx.y;                    // one head == one 64-col tile
  const int n0 = h * DK;

  v8f_t acc[2][4] = {};
  const int arow0 = m0 + (lane & 15);
  const int arow1 = m0 + 16 + (lane & 15);
  for (int kb = 0; kb < DMODEL; kb += 32) {
    if (kb + 32 < DMODEL) {  // prefetch next A k-slab (global_prefetch_b8)
      __builtin_prefetch(X + (size_t)arow0 * DMODEL + kb + 32, 0, 1);
      __builtin_prefetch(X + (size_t)arow1 * DMODEL + kb + 32, 0, 1);
    }
    v16bf_t a0 = frag_a_f32(X, arow0, DMODEL, kb, lane);
    v16bf_t a1 = frag_a_f32(X, arow1, DMODEL, kb, lane);
#pragma unroll
    for (int t = 0; t < 4; ++t) {
      v16bf_t bf = frag_b_bf16(Wbf, n0 + t * 16 + (lane & 15), DMODEL, kb, lane);
      acc[0][t] = wmma_bf16(a0, bf, acc[0][t]);
      acc[1][t] = wmma_bf16(a1, bf, acc[1][t]);
    }
  }

  const int b  = m0 / S_LEN;   // 256-row tiles never straddle a batch
  const int s0 = m0 % S_LEN;
#pragma unroll
  for (int mt = 0; mt < 2; ++mt) {
#pragma unroll
    for (int t = 0; t < 4; ++t) {
      const int dk = t * 16 + (lane & 15);
      const float bv = bias[n0 + dk];
#pragma unroll
      for (int r = 0; r < 8; ++r) {
        const int s = s0 + mt * 16 + ((lane & 16) ? 8 : 0) + r;
        const float val = acc[mt][t][r] + bv;
        const size_t idx =
            transposed ? (((size_t)(b * HEADS + h) * DK + dk) * S_LEN + s)
                       : (((size_t)(b * HEADS + h) * S_LEN + s) * DK + dk);
        out[idx] = (bf16_t)val;
      }
    }
  }
}

// ---- Kernel 2: flash-style attention, bf16 WMMA, fp32 softmax -------------
// grid = (16 q-blocks of 128 rows, 64 (b,h)), block = 256 (8 waves).
// Each wave owns a 16-row q tile; key blocks of 64 -> 16 WMMA per softmax.
__global__ __launch_bounds__(256) void attn_bf16_kernel(
    const bf16_t* __restrict__ Qw, const bf16_t* __restrict__ Kw,
    const bf16_t* __restrict__ Vt /* [B,H,Dk,S] */, float* __restrict__ out) {
  __shared__ bf16_t Pbuf[8][16][72];  // per-wave exp(P) tile, 144B row stride

  const int lane = threadIdx.x & 31;
  const int wave = threadIdx.x >> 5;
  const int m0 = blockIdx.x * 128 + wave * 16;  // q row tile
  const int bh = blockIdx.y;
  const int b = bh >> 4, h = bh & 15;

  const bf16_t* Qp = Qw + (size_t)bh * S_LEN * DK;
  const bf16_t* Kp = Kw + (size_t)bh * S_LEN * DK;
  const bf16_t* Vp = Vt + (size_t)bh * DK * S_LEN;

  // Q fragments for both 32-wide K-steps of the Dk=64 contraction.
  const int qrow = m0 + (lane & 15);
  const v16bf_t aq0 = frag_a_bf16(Qp, qrow, DK, 0, lane);
  const v16bf_t aq1 = frag_a_bf16(Qp, qrow, DK, 32, lane);

  v8f_t o[4] = {};
  float mi[8], li[8];
#pragma unroll
  for (int r = 0; r < 8; ++r) { mi[r] = -3.0e38f; li[r] = 0.0f; }

  bf16_t(*P)[72] = Pbuf[wave];
  const float scale = 0.125f;  // 1/sqrt(64)

  for (int j = 0; j < S_LEN; j += 64) {
    if (j + 64 < S_LEN) {  // prefetch next K/V slabs (global_prefetch_b8)
      __builtin_prefetch(Kp + (size_t)(j + 64 + (lane & 15)) * DK, 0, 1);
      __builtin_prefetch(Vp + (size_t)(lane & 15) * S_LEN + j + 64, 0, 1);
    }
    // scores tile S[16 x 64] = Q(16xDk) @ K^T
    v8f_t s[4] = {};
#pragma unroll
    for (int nt = 0; nt < 4; ++nt) {
      const int krow = j + nt * 16 + (lane & 15);
      v16bf_t bk0 = frag_b_bf16(Kp, krow, DK, 0, lane);
      s[nt] = wmma_bf16(aq0, bk0, s[nt]);
      v16bf_t bk1 = frag_b_bf16(Kp, krow, DK, 32, lane);
      s[nt] = wmma_bf16(aq1, bk1, s[nt]);
    }

    // online softmax per row; row r lives on one 16-lane half (ISA C layout)
#pragma unroll
    for (int r = 0; r < 8; ++r) {
      const float a0 = s[0][r] * scale, a1 = s[1][r] * scale;
      const float a2 = s[2][r] * scale, a3 = s[3][r] * scale;
      float mx = fmaxf(fmaxf(a0, a1), fmaxf(a2, a3));
      mx = fmaxf(mx, __shfl_xor(mx, 1));
      mx = fmaxf(mx, __shfl_xor(mx, 2));
      mx = fmaxf(mx, __shfl_xor(mx, 4));
      mx = fmaxf(mx, __shfl_xor(mx, 8));
      const float mnew  = fmaxf(mi[r], mx);
      const float alpha = __expf(mi[r] - mnew);
      mi[r] = mnew;
      const float p0 = __expf(a0 - mnew);
      const float p1 = __expf(a1 - mnew);
      const float p2 = __expf(a2 - mnew);
      const float p3 = __expf(a3 - mnew);
      float rs = (p0 + p1) + (p2 + p3);
      rs += __shfl_xor(rs, 1);
      rs += __shfl_xor(rs, 2);
      rs += __shfl_xor(rs, 4);
      rs += __shfl_xor(rs, 8);
      li[r] = li[r] * alpha + rs;
      o[0][r] *= alpha; o[1][r] *= alpha; o[2][r] *= alpha; o[3][r] *= alpha;
      const int row = ((lane & 16) ? 8 : 0) + r;
      P[row][(lane & 15)]      = (bf16_t)p0;
      P[row][16 + (lane & 15)] = (bf16_t)p1;
      P[row][32 + (lane & 15)] = (bf16_t)p2;
      P[row][48 + (lane & 15)] = (bf16_t)p3;
    }

    // O += P(16x64) @ V(64xDk); B^T rows are V^T rows (contiguous in S)
    const v16bf_t ap0 = frag_a_bf16(&P[0][0], lane & 15, 72, 0, lane);
    const v16bf_t ap1 = frag_a_bf16(&P[0][0], lane & 15, 72, 32, lane);
#pragma unroll
    for (int t = 0; t < 4; ++t) {
      const int vrow = t * 16 + (lane & 15);
      v16bf_t bv0 = frag_b_bf16(Vp, vrow, S_LEN, j, lane);
      o[t] = wmma_bf16(ap0, bv0, o[t]);
      v16bf_t bv1 = frag_b_bf16(Vp, vrow, S_LEN, j + 32, lane);
      o[t] = wmma_bf16(ap1, bv1, o[t]);
    }
  }

  // normalize + write fp32 output [B, S, DMODEL] merging heads
#pragma unroll
  for (int r = 0; r < 8; ++r) {
    const float inv = 1.0f / li[r];
    const int s = m0 + ((lane & 16) ? 8 : 0) + r;
    float* op = out + ((size_t)(b * S_LEN + s)) * DMODEL + h * DK + (lane & 15);
#pragma unroll
    for (int t = 0; t < 4; ++t) op[t * 16] = o[t][r] * inv;
  }
}

extern "C" void kernel_launch(void* const* d_in, const int* in_sizes, int n_in,
                              void* d_out, int out_size, void* d_ws,
                              size_t ws_size, hipStream_t stream) {
  const float* query = (const float*)d_in[0];
  const float* key_  = (const float*)d_in[1];
  const float* value = (const float*)d_in[2];
  const float* Wq = (const float*)d_in[3];
  const float* bq = (const float*)d_in[4];
  const float* Wk = (const float*)d_in[5];
  const float* bk = (const float*)d_in[6];
  const float* Wv = (const float*)d_in[7];
  const float* bv = (const float*)d_in[8];

  // Workspace: Q,K bf16 [B,H,S,Dk]; V bf16 transposed [B,H,Dk,S];
  // then 3 bf16 weight matrices. Total ~56.6 MB (L2-resident).
  const size_t per  = (size_t)4 * HEADS * S_LEN * DK;  // 8,388,608 elems
  const size_t wper = (size_t)DMODEL * DMODEL;         // 1,048,576 elems
  bf16_t* Qw  = (bf16_t*)d_ws;
  bf16_t* Kw  = Qw + per;
  bf16_t* Vw  = Kw + per;
  bf16_t* Wqb = Vw + per;
  bf16_t* Wkb = Wqb + wper;
  bf16_t* Wvb = Wkb + wper;

  dim3 block(256);

  dim3 gridc((unsigned)(wper / (256 * 8)));  // 512 blocks
  cvt_f32_bf16_kernel<<<gridc, block, 0, stream>>>(Wq, Wqb, (int)wper);
  cvt_f32_bf16_kernel<<<gridc, block, 0, stream>>>(Wk, Wkb, (int)wper);
  cvt_f32_bf16_kernel<<<gridc, block, 0, stream>>>(Wv, Wvb, (int)wper);

  dim3 grid1(32, 16);  // 8192/256 M-tiles x 16 heads
  proj_bf16_kernel<<<grid1, block, 0, stream>>>(query, Wqb, bq, Qw, 0);
  proj_bf16_kernel<<<grid1, block, 0, stream>>>(key_,  Wkb, bk, Kw, 0);
  proj_bf16_kernel<<<grid1, block, 0, stream>>>(value, Wvb, bv, Vw, 1);

  dim3 grid2(16, 64);  // 2048/128 q-blocks x (B*H)
  attn_bf16_kernel<<<grid2, block, 0, stream>>>(Qw, Kw, Vw, (float*)d_out);
}
